// LAS_83923660963855
// MI455X (gfx1250) — matrix operand order repaired and checked by reading, not compiled, not run on hardware
//
#include <hip/hip_runtime.h>
#include <hip/hip_bf16.h>

// ---------------------------------------------------------------------------
// Types
// ---------------------------------------------------------------------------
typedef __attribute__((ext_vector_type(16))) __bf16       bf16x16;
typedef __attribute__((ext_vector_type(8)))  float        f32x8;
typedef __attribute__((ext_vector_type(4)))  unsigned int u32x4;
typedef __attribute__((ext_vector_type(8)))  int          i32x8;
typedef __attribute__((ext_vector_type(4)))  int          i32x4;

union Frag {           // one WMMA operand fragment: 16 bf16 per lane = 32B
    bf16x16 v;
    u32x4   q[2];
    unsigned short u[16];
};

__device__ __forceinline__ __bf16 to_bf16(float f) {
    union { float f; unsigned u; } x; x.f = f;
    unsigned r = x.u + 0x7fffu + ((x.u >> 16) & 1u);     // RNE
    union { unsigned short s; __bf16 b; } y; y.s = (unsigned short)(r >> 16);
    return y.b;
}
__device__ __forceinline__ float from_bf16(__bf16 b) {
    union { unsigned short s; __bf16 b; } y; y.b = b;
    union { unsigned u; float f; } x; x.u = ((unsigned)y.s) << 16;
    return x.f;
}
__device__ __forceinline__ float sigm(float x) { return 1.f / (1.f + __expf(-x)); }

// ---------------------------------------------------------------------------
// WMMA fragment helpers.
//  A panel in LDS: row-major [64][lda] bf16.  16-bit A layout:
//  lane L holds row M=L&15, K runs {8*(L>>4).. +7} and {16+8*(L>>4).. +7}.
//  B is pre-packed frag-major: per 32-K chunk, per 16-N tile, 32 lanes x 16
//  contiguous bf16 (lane L covers N=L&15, K={8*(L>>4)..+7},{16+8*(L>>4)..+7}, as A).
// ---------------------------------------------------------------------------
__device__ __forceinline__ void load_a_frag(const __bf16* A0, int lda, int mt, Frag& f) {
    int lane = threadIdx.x & 31;
    const __bf16* p = A0 + (size_t)(mt * 16 + (lane & 15)) * lda + ((lane >> 4) << 3);
    f.q[0] = *(const u32x4*)p;
    f.q[1] = *(const u32x4*)(p + 16);
}
__device__ __forceinline__ void load_b_frag(const __bf16* Bp, int nt, Frag& f) {
    int lane = threadIdx.x & 31;
    const u32x4* p = (const u32x4*)(Bp + (((size_t)nt * 32 + lane) << 4));
    f.q[0] = p[0];
    f.q[1] = p[1];
}
__device__ __forceinline__ f32x8 wmma_bf16(const Frag& a, const Frag& b, f32x8 c) {
    return __builtin_amdgcn_wmma_f32_16x16x32_bf16(
        /*neg_a=*/false, a.v, /*neg_b=*/false, b.v,
        /*c_mod=*/(short)0, c, /*reuse_a=*/false, /*reuse_b=*/false);
}

// ---------------------------------------------------------------------------
// TDM: copy `bytes` (multiple of 8) from global to LDS as a 1-D tensor tile.
// D# built per CDNA5 ISA ch.8 (group0: count/lds/global/type, group1: dims).
// 6-arg builtin form (clang-23 / therock-10.0): (g0, g1, g2, g3, g4, cpol).
// Issued by one lane; completion via s_wait_tensorcnt.
// ---------------------------------------------------------------------------
__device__ __forceinline__ void tdm_copy_to_lds(void* lds, const void* g, unsigned bytes) {
    unsigned units = bytes >> 3;                               // 8-byte elements
    unsigned long long ga = (unsigned long long)(size_t)g;
    unsigned la = (unsigned)(size_t)lds;                       // LDS byte offset
    u32x4 g0 = { 1u,                                           // count=1 (user D#)
                 la,                                           // lds_addr
                 (unsigned)ga,                                 // global_addr[31:0]
                 ((unsigned)(ga >> 32) & 0x01ffffffu) | 0x80000000u }; // addr msbs | type=2
    i32x8 g1 = { (int)(3u << 16),                              // data_size=3 (8B)
                 (int)((units & 0xffffu) << 16),               // tensor_dim0[15:0]
                 (int)(((units >> 16) & 0xffffu) | (1u << 16)),// dim0[31:16] | tensor_dim1=1
                 (int)((units & 0xffffu) << 16),               // tile_dim0
                 0,                                            // tile_dim1/2 unused
                 (int)units,                                   // tensor_dim0_stride[31:0]
                 0, 0 };
    i32x4 gz4 = { 0, 0, 0, 0 };
    i32x8 gz8 = { 0, 0, 0, 0, 0, 0, 0, 0 };
    __builtin_amdgcn_tensor_load_to_lds(g0, g1, gz4, gz4, gz8, 0);
}

// ---------------------------------------------------------------------------
// Fused WMMA LSTM cell step for a 64-batch, H=128 cell (gates N=512).
// Wave w owns gate tiles {w, w+8, w+16, w+24} -> lane-local i,f,g,o.
// K-loop is split by address space: chunks [0,kcSplit) stream from global
// packed weights (global_load_b128, LOADcnt only); chunks [kcSplit,nkc)
// read the LDS-resident Whh copy (ds_load_b128, DScnt only).  Keeping the
// two pointers un-merged avoids flat_load + combined-counter waits.
// ---------------------------------------------------------------------------
__device__ __forceinline__ void run_cell(
    const __bf16* A, int lda, int Ktot, int kcSplit,
    const __bf16* W0, const __bf16* W1,
    const float* bih, const float* bhh, float* c,
    __bf16* hA, int ldhA, int offA,          // recurrent bf16 h slot
    __bf16* hB, int ldhB, int offB,          // optional second bf16 dest
    float* hF, int ldhF,                     // optional fp32 h out
    float* cF, int ldcF)                     // optional fp32 c out
{
    int tid = threadIdx.x, wv = tid >> 5, lane = tid & 31;
    int jn = wv * 16 + (lane & 15);
    float bi = bih[jn]       + bhh[jn];
    float bf = bih[jn + 128] + bhh[jn + 128];
    float bg = bih[jn + 256] + bhh[jn + 256];
    float bo = bih[jn + 384] + bhh[jn + 384];
    float hout[32];
    int nkc = Ktot >> 5;
    for (int mt = 0; mt < 4; ++mt) {
        f32x8 ai = {}, af = {}, ag = {}, ao = {};
        for (int kc = 0; kc < kcSplit; ++kc) {                   // global weights
            const __bf16* Bp = W0 + (size_t)kc * 16384;
            Frag a, b;
            load_a_frag(A + kc * 32, lda, mt, a);
            load_b_frag(Bp, wv,      b); ai = wmma_bf16(a, b, ai);
            load_b_frag(Bp, wv + 8,  b); af = wmma_bf16(a, b, af);
            load_b_frag(Bp, wv + 16, b); ag = wmma_bf16(a, b, ag);
            load_b_frag(Bp, wv + 24, b); ao = wmma_bf16(a, b, ao);
        }
        for (int kc = kcSplit; kc < nkc; ++kc) {                 // LDS weights
            const __bf16* Bp = W1 + (size_t)(kc - kcSplit) * 16384;
            Frag a, b;
            load_a_frag(A + kc * 32, lda, mt, a);
            load_b_frag(Bp, wv,      b); ai = wmma_bf16(a, b, ai);
            load_b_frag(Bp, wv + 8,  b); af = wmma_bf16(a, b, af);
            load_b_frag(Bp, wv + 16, b); ag = wmma_bf16(a, b, ag);
            load_b_frag(Bp, wv + 24, b); ao = wmma_bf16(a, b, ao);
        }
#pragma unroll
        for (int v = 0; v < 8; ++v) {
            int idx = mt * 8 + v;
            float ig = sigm(ai[v] + bi);
            float fg = sigm(af[v] + bf);
            float gg = tanhf(ag[v] + bg);
            float og = sigm(ao[v] + bo);
            float cc = fg * c[idx] + ig * gg;
            c[idx] = cc;
            hout[idx] = og * tanhf(cc);
        }
    }
    __syncthreads();                 // everyone done reading old h
    for (int mt = 0; mt < 4; ++mt) {
#pragma unroll
        for (int v = 0; v < 8; ++v) {
            int brow = mt * 16 + ((lane >> 4) << 3) + v;
            float h = hout[mt * 8 + v];
            __bf16 hb = to_bf16(h);
            hA[brow * ldhA + offA + jn] = hb;
            if (hB) hB[brow * ldhB + offB + jn] = hb;
            if (hF) hF[(long long)brow * ldhF + jn] = h;
            if (cF) cF[brow * ldcF + jn] = c[mt * 8 + v];
        }
    }
    __syncthreads();
}

// ---------------------------------------------------------------------------
// Weight packer: fp32 -> bf16 WMMA-B fragment order.
// mode 0: linear W[N][K];  mode 1: conv W[N][C][2], k = kk*C + c.
// Layout: [kc][nt][lane(32)][e(16)], zero-padded for k>=K / n>=N.
// ---------------------------------------------------------------------------
__global__ void pack_b(const float* __restrict__ W, __bf16* __restrict__ dst,
                       int N, int K, int Kpad, int nT, int C, int mode) {
    long long total = (long long)(Kpad >> 5) * nT * 512;
    for (long long idx = (long long)blockIdx.x * 256 + threadIdx.x; idx < total;
         idx += (long long)gridDim.x * 256) {
        int e = (int)(idx & 15);
        int lane = (int)((idx >> 4) & 31);
        int nt = (int)((idx >> 9) % nT);
        int kc = (int)(idx / ((long long)512 * nT));
        int k = (kc << 5) + ((lane >> 4) << 3) + (e & 7) + ((e >> 3) << 4);
        int n = (nt << 4) + (lane & 15);
        float v = 0.f;
        if (k < K && n < N) {
            if (mode == 0) v = W[(long long)n * K + k];
            else { int kk = k / C, cc = k - kk * C; v = W[((long long)n * C + cc) * 2 + kk]; }
        }
        dst[idx] = to_bf16(v);
    }
}

__global__ void zero_f32(float* p, int n) {
    int i = blockIdx.x * 256 + threadIdx.x;
    if (i < n) p[i] = 0.f;
}

__global__ void bn_finalize(const float* __restrict__ s, const float* __restrict__ q,
                            float cnt, const float* __restrict__ g,
                            const float* __restrict__ b,
                            float* __restrict__ scale, float* __restrict__ shift) {
    int c = threadIdx.x;                       // 128 threads
    float m = s[c] / cnt;
    float v = q[c] / cnt - m * m;
    float sc = g[c] * rsqrtf(v + 1e-5f);
    scale[c] = sc;
    shift[c] = b[c] - m * sc;
}

// ---------------------------------------------------------------------------
// Persistent LSTM over a sequence; grid.x in {1,2} (dir 1 = reversed, writes
// column offset 128 of the concat buffer). A-panel [64][Ktot] bf16 in LDS;
// Whh packed chunks copied to LDS once via TDM; Wih streamed from L2.
// ---------------------------------------------------------------------------
__global__ void __launch_bounds__(256) lstm_seq(
    const float* __restrict__ X, long long xts, long long xbs, int I, int Kpad,
    const float* __restrict__ scale, const float* __restrict__ shift,
    const __bf16* __restrict__ Wp0, const float* __restrict__ bih0, const float* __restrict__ bhh0,
    const __bf16* __restrict__ Wp1, const float* __restrict__ bih1, const float* __restrict__ bhh1,
    float* __restrict__ Hout, long long hts, int hbs, int T)
{
    int dir = blockIdx.x;
    const __bf16* Wp = dir ? Wp1 : Wp0;
    const float* bih = dir ? bih1 : bih0;
    const float* bhh = dir ? bhh1 : bhh0;
    int hoff = dir * 128;
    int Ktot = Kpad + 128;
    int kcI = Kpad >> 5;
    extern __shared__ __align__(16) char smem[];
    __bf16* A  = (__bf16*)smem;                                  // [64][Ktot]
    __bf16* Wl = (__bf16*)(smem + (size_t)64 * Ktot * 2);        // 4 chunks = 128KB
    int tid = threadIdx.x;
    if (tid == 0) tdm_copy_to_lds(Wl, Wp + (size_t)kcI * 16384, 4u * 16384u * 2u);
    for (int idx = tid; idx < 64 * 128; idx += 256)              // zero h0
        A[(idx >> 7) * Ktot + Kpad + (idx & 127)] = to_bf16(0.f);
    __builtin_amdgcn_s_wait_tensorcnt(0);
    __syncthreads();
    float c[32];
#pragma unroll
    for (int i = 0; i < 32; ++i) c[i] = 0.f;

    for (int t = 0; t < T; ++t) {
        int tp = dir ? (T - 1 - t) : t;
        const float* xt = X + (long long)tp * xts;
        for (int idx = tid; idx < 64 * Kpad; idx += 256) {       // stage x_t (+affine)
            int b = idx / Kpad, k = idx - b * Kpad;
            float v = 0.f;
            if (k < I) {
                v = xt[(long long)b * xbs + k];
                if (scale) v = v * scale[k] + shift[k];
            }
            A[b * Ktot + k] = to_bf16(v);
        }
        if (t + 1 < T) {
            int tn = dir ? (T - 2 - t) : (t + 1);
            __builtin_prefetch((const void*)(X + (long long)tn * xts + tid), 0, 1);
        }
        __syncthreads();
        run_cell(A, Ktot, Ktot, kcI, Wp, Wl, bih, bhh, c,
                 A, Ktot, Kpad, nullptr, 0, 0,
                 Hout + (long long)tp * hts + hoff, hbs, nullptr, 0);
    }
}

// ---------------------------------------------------------------------------
// Row GEMM (N=128) for conv(k=2,s=2) as GEMM and for the K/V projections.
// mode 1: A-row r=(to*64+b), A[k]=x[(2to+kk)*64+b][c] with k=kk*C+c.
// Optional fused BatchNorm batch-stat accumulation via atomics.
// ---------------------------------------------------------------------------
__global__ void __launch_bounds__(256) gemm_rows(
    const float* __restrict__ X, int mode, int C, int K, int Kpad,
    const __bf16* __restrict__ Wp, const float* __restrict__ bias,
    float* __restrict__ Out, float* bnsum, float* bnsq)
{
    extern __shared__ __align__(16) char smem[];
    __bf16* A = (__bf16*)smem;                                   // [64][Kpad]
    int tid = threadIdx.x;
    long long r0 = (long long)blockIdx.x * 64;
    for (int idx = tid; idx < 64 * Kpad; idx += 256) {
        int br = idx / Kpad, k = idx - br * Kpad;
        float v = 0.f;
        if (k < K) {
            long long r = r0 + br;
            if (mode == 0) v = X[r * K + k];
            else {
                int to = (int)(r >> 6), b = (int)(r & 63);
                int kk = k / C, cc = k - kk * C;
                v = X[((long long)(2 * to + kk) * 64 + b) * C + cc];
            }
        }
        A[br * Kpad + k] = to_bf16(v);
    }
    __syncthreads();
    int wv = tid >> 5, lane = tid & 31;
    int col = wv * 16 + (lane & 15);
    float bc = bias ? bias[col] : 0.f;
    float s = 0.f, ss = 0.f;
    int nkc = Kpad >> 5;
    for (int mt = 0; mt < 4; ++mt) {
        f32x8 acc = {};
        for (int kc = 0; kc < nkc; ++kc) {
            Frag a, b;
            load_a_frag(A + kc * 32, Kpad, mt, a);
            load_b_frag(Wp + (size_t)kc * 4096, wv, b);          // nT=8 chunk stride
            acc = wmma_bf16(a, b, acc);
        }
#pragma unroll
        for (int v = 0; v < 8; ++v) {
            float y = acc[v] + bc;
            int brow = mt * 16 + ((lane >> 4) << 3) + v;
            Out[(r0 + brow) * 128 + col] = y;
            s += y; ss += y * y;
        }
    }
    if (bnsum) { atomicAdd(&bnsum[col], s); atomicAdd(&bnsq[col], ss); }
}

// ---------------------------------------------------------------------------
// Persistent attention decoder (speller): 255 steps, one WGP.
// LDS carve (312320 B): A0[64][768] | A1[64][256] | A2 | AL | c2f(f32)[64][128]
//   | probs(f32)[64][256] (aliased by obuf) | A3[64][128] | red[2][128].
// ---------------------------------------------------------------------------
__global__ void __launch_bounds__(256) speller(
    const float* __restrict__ hk, const float* __restrict__ hv,   // (256,64,128)
    const int* __restrict__ targets, const float* __restrict__ emb,
    const __bf16* __restrict__ Wc0, const float* __restrict__ b0i, const float* __restrict__ b0h,
    const __bf16* __restrict__ Wc1, const float* __restrict__ b1i, const float* __restrict__ b1h,
    const __bf16* __restrict__ Wc2, const float* __restrict__ b2i, const float* __restrict__ b2h,
    const __bf16* __restrict__ Wl12, const float* __restrict__ bl1, const float* __restrict__ bl2,
    const float* __restrict__ l3W, const float* __restrict__ l3b,
    const float* __restrict__ sbng, const float* __restrict__ sbnb,
    const float* __restrict__ sc0, const float* __restrict__ sc1, const float* __restrict__ sc2,
    const float* __restrict__ sh0, const float* __restrict__ sh1, const float* __restrict__ sh2,
    float* __restrict__ preds)                                    // (64,255,34)
{
    extern __shared__ __align__(16) char smem[];
    __bf16* A0 = (__bf16*)(smem);
    __bf16* A1 = (__bf16*)(smem + 98304);
    __bf16* A2 = (__bf16*)(smem + 131072);
    __bf16* AL = (__bf16*)(smem + 163840);
    float*  c2f = (float*)(smem + 196608);
    float*  probs = (float*)(smem + 229376);
    float*  obuf = probs;                       // alias: probs dead before lin
    __bf16* A3 = (__bf16*)(smem + 294912);
    float*  red = (float*)(smem + 311296);
    float*  logits = c2f;                       // alias: c2f dead after attention

    int tid = threadIdx.x, wv = tid >> 5, lane = tid & 31;
    int jn = wv * 16 + (lane & 15);

    float c0[32], c1[32], c2[32];
#pragma unroll
    for (int mt = 0; mt < 4; ++mt)
#pragma unroll
        for (int v = 0; v < 8; ++v) {
            int brow = mt * 16 + ((lane >> 4) << 3) + v;
            c0[mt * 8 + v] = sc0[brow * 128 + jn];
            c1[mt * 8 + v] = sc1[brow * 128 + jn];
            c2[mt * 8 + v] = sc2[brow * 128 + jn];
        }
    for (int idx = tid; idx < 64 * 128; idx += 256) {
        int b = idx >> 7, j = idx & 127;
        A0[b * 768 + 512 + j] = to_bf16(0.f);            // ctx_{-1} = 0
        A0[b * 768 + 640 + j] = to_bf16(sh0[idx]);
        A1[b * 256 + 128 + j] = to_bf16(sh1[idx]);
        A2[b * 256 + 128 + j] = to_bf16(sh2[idx]);
    }
    __syncthreads();

    for (int t = 0; t < 255; ++t) {
        for (int idx = tid; idx < 64 * 512; idx += 256) {        // stage y_emb
            int b = idx >> 9, k = idx & 511;
            int tg = targets[b * 256 + t];
            A0[b * 768 + k] = to_bf16(emb[tg * 512 + k]);
        }
        __syncthreads();
        run_cell(A0, 768, 768, 24, Wc0, nullptr, b0i, b0h, c0,
                 A0, 768, 640, A1, 256, 0, nullptr, 0, nullptr, 0);
        run_cell(A1, 256, 256, 8, Wc1, nullptr, b1i, b1h, c1,
                 A1, 256, 128, A2, 256, 0, nullptr, 0, nullptr, 0);
        run_cell(A2, 256, 256, 8, Wc2, nullptr, b2i, b2h, c2,
                 A2, 256, 128, AL, 256, 0, nullptr, 0, c2f, 128);

        // attention scores + softmax (wave wv -> batch rows wv*8..wv*8+7)
        for (int bb = 0; bb < 8; ++bb) {
            int b = wv * 8 + bb;
            float sc[8];
            float mx = -1e30f;
            const float* cp = c2f + b * 128;
#pragma unroll
            for (int q = 0; q < 8; ++q) {
                int s = q * 32 + lane;
                const float* kp = hk + ((long long)s * 64 + b) * 128;
                float d = 0.f;
                for (int o = 0; o < 128; ++o) d += kp[o] * cp[o];
                sc[q] = d; mx = fmaxf(mx, d);
            }
            for (int m = 16; m >= 1; m >>= 1) mx = fmaxf(mx, __shfl_xor(mx, m, 32));
            float sum = 0.f;
#pragma unroll
            for (int q = 0; q < 8; ++q) { sc[q] = __expf(sc[q] - mx); sum += sc[q]; }
            for (int m = 16; m >= 1; m >>= 1) sum += __shfl_xor(sum, m, 32);
            float inv = 1.f / sum;
#pragma unroll
            for (int q = 0; q < 8; ++q) probs[b * 256 + q * 32 + lane] = sc[q] * inv;
        }
        __syncthreads();
        // context
        for (int bb = 0; bb < 8; ++bb) {
            int b = wv * 8 + bb;
#pragma unroll
            for (int p = 0; p < 4; ++p) {
                int o = p * 32 + lane;
                float acc = 0.f;
                for (int s = 0; s < 256; ++s)
                    acc += probs[b * 256 + s] * hv[((long long)s * 64 + b) * 128 + o];
                __bf16 cb = to_bf16(acc);
                A0[b * 768 + 512 + o] = cb;              // feed next step's cell0
                AL[b * 256 + 128 + o] = cb;              // feed lin2
            }
        }
        __syncthreads();
        // o = [h2|ctx] @ [W1;W2]^T + b1 + b2   (WMMA, N=128)
        {
            float bc = bl1[jn] + bl2[jn];
            for (int mt = 0; mt < 4; ++mt) {
                f32x8 acc = {};
                for (int kc = 0; kc < 8; ++kc) {
                    Frag a, b;
                    load_a_frag(AL + kc * 32, 256, mt, a);
                    load_b_frag(Wl12 + (size_t)kc * 4096, wv, b);
                    acc = wmma_bf16(a, b, acc);
                }
#pragma unroll
                for (int v = 0; v < 8; ++v) {
                    int brow = mt * 16 + ((lane >> 4) << 3) + v;
                    obuf[brow * 128 + jn] = acc[v] + bc;
                }
            }
        }
        __syncthreads();
        // per-step BatchNorm over batch
        if (tid < 128) {
            float s = 0.f, ss = 0.f;
            for (int b = 0; b < 64; ++b) { float x = obuf[b * 128 + tid]; s += x; ss += x * x; }
            float m = s * (1.f / 64.f), vv = ss * (1.f / 64.f) - m * m;
            float scl = sbng[tid] * rsqrtf(vv + 1e-5f);
            red[tid] = scl; red[128 + tid] = sbnb[tid] - m * scl;
        }
        __syncthreads();
        for (int idx = tid; idx < 64 * 128; idx += 256) {
            int j = idx & 127;
            float y = obuf[idx] * red[j] + red[128 + j];
            A3[idx] = to_bf16(fmaxf(y, 0.f));
        }
        __syncthreads();
        // logits (V=34) + softmax
        for (int idx = tid; idx < 64 * 34; idx += 256) {
            int b = idx / 34, vv2 = idx - b * 34;
            const float* w = l3W + vv2 * 128;
            float d = l3b[vv2];
            for (int k2 = 0; k2 < 128; ++k2) d += from_bf16(A3[b * 128 + k2]) * w[k2];
            logits[b * 34 + vv2] = d;
        }
        __syncthreads();
        if (tid < 64) {
            int b = tid;
            float mx = -1e30f;
            for (int v2 = 0; v2 < 34; ++v2) mx = fmaxf(mx, logits[b * 34 + v2]);
            float sum = 0.f, e[34];
            for (int v2 = 0; v2 < 34; ++v2) { e[v2] = __expf(logits[b * 34 + v2] - mx); sum += e[v2]; }
            float inv = 1.f / sum;
            for (int v2 = 0; v2 < 34; ++v2)
                preds[((long long)b * 255 + t) * 34 + v2] = e[v2] * inv;
        }
        __syncthreads();
    }
}

// ---------------------------------------------------------------------------
// Host orchestration.
// Input order assumes pytree alphabetical flatten: params(*) < targets < utter,
// with sorted keys at each dict level ('Whh'<'Wih'<'bhh'<'bih', 'W'<'b', ...).
// ---------------------------------------------------------------------------
enum {
    IN_C0_WHH = 0, IN_C0_WIH, IN_C0_BHH, IN_C0_BIH,
    IN_C1_WHH, IN_C1_WIH, IN_C1_BHH, IN_C1_BIH,
    IN_C2_WHH, IN_C2_WIH, IN_C2_BHH, IN_C2_BIH,
    IN_CONV1_W, IN_CONV1_B, IN_CONV2_W, IN_CONV2_B, IN_CONV3_W, IN_CONV3_B,
    IN_EMB,
    IN_SC0, IN_SC1, IN_SC2, IN_SH0, IN_SH1, IN_SH2,
    IN_KLIN_W, IN_KLIN_B,
    IN_LBN_B, IN_LBN_G,
    IN_LIN1_W, IN_LIN1_B, IN_LIN2_W, IN_LIN2_B, IN_LIN3_W, IN_LIN3_B,
    IN_RNN1_WHH, IN_RNN1_WIH, IN_RNN1_BHH, IN_RNN1_BIH,
    IN_RNN2_WHH, IN_RNN2_WIH, IN_RNN2_BHH, IN_RNN2_BIH,
    IN_RNN3_WHH, IN_RNN3_WIH, IN_RNN3_BHH, IN_RNN3_BIH,
    IN_RNNF_WHH, IN_RNNF_WIH, IN_RNNF_BHH, IN_RNNF_BIH,
    IN_RNNR_WHH, IN_RNNR_WIH, IN_RNNR_BHH, IN_RNNR_BIH,
    IN_SBN_B, IN_SBN_G,
    IN_VLIN_W, IN_VLIN_B,
    IN_TARGETS, IN_UTTER
};

extern "C" void kernel_launch(void* const* d_in, const int* in_sizes, int n_in,
                              void* d_out, int out_size, void* d_ws, size_t ws_size,
                              hipStream_t stream) {
    (void)in_sizes; (void)n_in; (void)out_size; (void)ws_size;
#define FIN(i) ((const float*)d_in[(i)])

    char* wsp = (char*)d_ws;
    size_t off = 0;
    auto alloc = [&](size_t bytes) -> char* {
        char* p = wsp + off;
        off = (off + bytes + 255) & ~(size_t)255;
        return p;
    };

    // packed bf16 weights
    __bf16* pk_rnnf = (__bf16*)alloc((size_t)98304 * 2);
    __bf16* pk_rnnr = (__bf16*)alloc((size_t)98304 * 2);
    __bf16* pk_rnn1 = (__bf16*)alloc((size_t)131072 * 2);
    __bf16* pk_rnn2 = (__bf16*)alloc((size_t)131072 * 2);
    __bf16* pk_rnn3 = (__bf16*)alloc((size_t)131072 * 2);
    __bf16* pk_c0   = (__bf16*)alloc((size_t)393216 * 2);
    __bf16* pk_c1   = (__bf16*)alloc((size_t)131072 * 2);
    __bf16* pk_c2   = (__bf16*)alloc((size_t)131072 * 2);
    __bf16* pk_cv1  = (__bf16*)alloc((size_t)65536 * 2);
    __bf16* pk_cv2  = (__bf16*)alloc((size_t)32768 * 2);
    __bf16* pk_cv3  = (__bf16*)alloc((size_t)32768 * 2);
    __bf16* pk_k    = (__bf16*)alloc((size_t)16384 * 2);
    __bf16* pk_v    = (__bf16*)alloc((size_t)16384 * 2);
    __bf16* pk_l12  = (__bf16*)alloc((size_t)32768 * 2);

    // fp32 sequence buffers (aliased across pyramid stages)
    float* BUF0 = (float*)alloc((size_t)2048 * 64 * 256 * 4);   // hcat / y2 / h3
    float* BUF1 = (float*)alloc((size_t)1024 * 64 * 128 * 4);   // y1 / h2
    float* BUF2 = (float*)alloc((size_t)1024 * 64 * 128 * 4);   // h1 / y3
    float* HK   = (float*)alloc((size_t)256 * 64 * 128 * 4);
    float* HV   = (float*)alloc((size_t)256 * 64 * 128 * 4);
    float* BNS  = (float*)alloc(256 * 4);   float* BNQ = BNS + 128;
    float* SCL  = (float*)alloc(256 * 4);   float* SHF = SCL + 128;

    (void)hipFuncSetAttribute((const void*)lstm_seq,
        hipFuncAttributeMaxDynamicSharedMemorySize, 163840);
    (void)hipFuncSetAttribute((const void*)gemm_rows,
        hipFuncAttributeMaxDynamicSharedMemorySize, 65536);
    (void)hipFuncSetAttribute((const void*)speller,
        hipFuncAttributeMaxDynamicSharedMemorySize, 312320);

    auto pack = [&](const float* W, __bf16* dst, int N, int K, int Kpad, int nT, int C, int mode) {
        long long total = (long long)(Kpad >> 5) * nT * 512;
        int g = (int)((total + 255) / 256);
        pack_b<<<dim3(g), dim3(256), 0, stream>>>(W, dst, N, K, Kpad, nT, C, mode);
    };

    // ---- pack all weights (B-frag order, [Wih chunks | Whh chunks]) ----
    pack(FIN(IN_RNNF_WIH), pk_rnnf,              512, 40, 64, 32, 0, 0);
    pack(FIN(IN_RNNF_WHH), pk_rnnf + 2 * 16384,  512, 128, 128, 32, 0, 0);
    pack(FIN(IN_RNNR_WIH), pk_rnnr,              512, 40, 64, 32, 0, 0);
    pack(FIN(IN_RNNR_WHH), pk_rnnr + 2 * 16384,  512, 128, 128, 32, 0, 0);
    pack(FIN(IN_RNN1_WIH), pk_rnn1,              512, 128, 128, 32, 0, 0);
    pack(FIN(IN_RNN1_WHH), pk_rnn1 + 4 * 16384,  512, 128, 128, 32, 0, 0);
    pack(FIN(IN_RNN2_WIH), pk_rnn2,              512, 128, 128, 32, 0, 0);
    pack(FIN(IN_RNN2_WHH), pk_rnn2 + 4 * 16384,  512, 128, 128, 32, 0, 0);
    pack(FIN(IN_RNN3_WIH), pk_rnn3,              512, 128, 128, 32, 0, 0);
    pack(FIN(IN_RNN3_WHH), pk_rnn3 + 4 * 16384,  512, 128, 128, 32, 0, 0);
    pack(FIN(IN_C0_WIH),   pk_c0,                512, 640, 640, 32, 0, 0);
    pack(FIN(IN_C0_WHH),   pk_c0 + 20 * 16384,   512, 128, 128, 32, 0, 0);
    pack(FIN(IN_C1_WIH),   pk_c1,                512, 128, 128, 32, 0, 0);
    pack(FIN(IN_C1_WHH),   pk_c1 + 4 * 16384,    512, 128, 128, 32, 0, 0);
    pack(FIN(IN_C2_WIH),   pk_c2,                512, 128, 128, 32, 0, 0);
    pack(FIN(IN_C2_WHH),   pk_c2 + 4 * 16384,    512, 128, 128, 32, 0, 0);
    pack(FIN(IN_CONV1_W),  pk_cv1, 128, 512, 512, 8, 256, 1);
    pack(FIN(IN_CONV2_W),  pk_cv2, 128, 256, 256, 8, 128, 1);
    pack(FIN(IN_CONV3_W),  pk_cv3, 128, 256, 256, 8, 128, 1);
    pack(FIN(IN_KLIN_W),   pk_k,   128, 128, 128, 8, 0, 0);
    pack(FIN(IN_VLIN_W),   pk_v,   128, 128, 128, 8, 0, 0);
    pack(FIN(IN_LIN1_W),   pk_l12,             128, 128, 128, 8, 0, 0);
    pack(FIN(IN_LIN2_W),   pk_l12 + 4 * 4096,  128, 128, 128, 8, 0, 0);

    // ---- bottom bidirectional LSTM: both directions in parallel (grid=2) ----
    lstm_seq<<<dim3(2), dim3(256), 64 * 192 * 2 + 131072, stream>>>(
        FIN(IN_UTTER), 40LL, 81920LL, 40, 64, nullptr, nullptr,
        pk_rnnf, FIN(IN_RNNF_BIH), FIN(IN_RNNF_BHH),
        pk_rnnr, FIN(IN_RNNR_BIH), FIN(IN_RNNR_BHH),
        BUF0, 16384LL, 256, 2048);

    // ---- pyramid level 1: conv+BN (GEMM) then LSTM ----
    zero_f32<<<dim3(1), dim3(256), 0, stream>>>(BNS, 256);
    gemm_rows<<<dim3(1024), dim3(256), 64 * 512 * 2, stream>>>(
        BUF0, 1, 256, 512, 512, pk_cv1, FIN(IN_CONV1_B), BUF1, BNS, BNQ);
    bn_finalize<<<dim3(1), dim3(128), 0, stream>>>(BNS, BNQ, 65536.f,
        FIN(IN_LBN_G), FIN(IN_LBN_B), SCL, SHF);
    lstm_seq<<<dim3(1), dim3(256), 64 * 256 * 2 + 131072, stream>>>(
        BUF1, 8192LL, 128LL, 128, 128, SCL, SHF,
        pk_rnn1, FIN(IN_RNN1_BIH), FIN(IN_RNN1_BHH),
        pk_rnn1, FIN(IN_RNN1_BIH), FIN(IN_RNN1_BHH),
        BUF2, 8192LL, 128, 1024);

    // ---- pyramid level 2 ----
    zero_f32<<<dim3(1), dim3(256), 0, stream>>>(BNS, 256);
    gemm_rows<<<dim3(512), dim3(256), 64 * 256 * 2, stream>>>(
        BUF2, 1, 128, 256, 256, pk_cv2, FIN(IN_CONV2_B), BUF0, BNS, BNQ);
    bn_finalize<<<dim3(1), dim3(128), 0, stream>>>(BNS, BNQ, 32768.f,
        FIN(IN_LBN_G), FIN(IN_LBN_B), SCL, SHF);
    lstm_seq<<<dim3(1), dim3(256), 64 * 256 * 2 + 131072, stream>>>(
        BUF0, 8192LL, 128LL, 128, 128, SCL, SHF,
        pk_rnn2, FIN(IN_RNN2_BIH), FIN(IN_RNN2_BHH),
        pk_rnn2, FIN(IN_RNN2_BIH), FIN(IN_RNN2_BHH),
        BUF1, 8192LL, 128, 512);

    // ---- pyramid level 3 ----
    zero_f32<<<dim3(1), dim3(256), 0, stream>>>(BNS, 256);
    gemm_rows<<<dim3(256), dim3(256), 64 * 256 * 2, stream>>>(
        BUF1, 1, 128, 256, 256, pk_cv3, FIN(IN_CONV3_B), BUF2, BNS, BNQ);
    bn_finalize<<<dim3(1), dim3(128), 0, stream>>>(BNS, BNQ, 16384.f,
        FIN(IN_LBN_G), FIN(IN_LBN_B), SCL, SHF);
    lstm_seq<<<dim3(1), dim3(256), 64 * 256 * 2 + 131072, stream>>>(
        BUF2, 8192LL, 128LL, 128, 128, SCL, SHF,
        pk_rnn3, FIN(IN_RNN3_BIH), FIN(IN_RNN3_BHH),
        pk_rnn3, FIN(IN_RNN3_BIH), FIN(IN_RNN3_BHH),
        BUF0, 8192LL, 128, 256);

    // ---- key/value projections ----
    gemm_rows<<<dim3(256), dim3(256), 64 * 128 * 2, stream>>>(
        BUF0, 0, 0, 128, 128, pk_k, FIN(IN_KLIN_B), HK, nullptr, nullptr);
    gemm_rows<<<dim3(256), dim3(256), 64 * 128 * 2, stream>>>(
        BUF0, 0, 0, 128, 128, pk_v, FIN(IN_VLIN_B), HV, nullptr, nullptr);

    // ---- attention decoder ----
    speller<<<dim3(1), dim3(256), 312320, stream>>>(
        HK, HV, (const int*)d_in[IN_TARGETS], FIN(IN_EMB),
        pk_c0, FIN(IN_C0_BIH), FIN(IN_C0_BHH),
        pk_c1, FIN(IN_C1_BIH), FIN(IN_C1_BHH),
        pk_c2, FIN(IN_C2_BIH), FIN(IN_C2_BHH),
        pk_l12, FIN(IN_LIN1_B), FIN(IN_LIN2_B),
        FIN(IN_LIN3_W), FIN(IN_LIN3_B),
        FIN(IN_SBN_G), FIN(IN_SBN_B),
        FIN(IN_SC0), FIN(IN_SC1), FIN(IN_SC2),
        FIN(IN_SH0), FIN(IN_SH1), FIN(IN_SH2),
        (float*)d_out);
#undef FIN
}
